// JadioAttention_28303834480645
// MI455X (gfx1250) — matrix operand order repaired and checked by reading, not compiled, not run on hardware
//
#include <hip/hip_runtime.h>

// Problem constants (reference: B=4, S=2048, H=2048, NH=16, HD=128)
#define H_    2048
#define NH_   16
#define HD_   128
#define BATCH 4
#define SEQ   2048
#define M_TOT (BATCH * SEQ)   // 8192
#define N_QKV (3 * H_)        // 6144
#define SCALE_ 0.08838834764831845f   // 1/sqrt(128)

typedef __attribute__((ext_vector_type(16))) __bf16 v16bf;
typedef __attribute__((ext_vector_type(8)))  float  v8f;

union FragAB { v16bf v; unsigned int u[8]; uint4 q[2]; };
union AccU   { v8f v; float f[8]; };

// float -> bf16 bits, round-to-nearest-even
__device__ __forceinline__ unsigned short f2bf(float f) {
  unsigned int u = __float_as_uint(f);
  u += 0x7FFFu + ((u >> 16) & 1u);
  return (unsigned short)(u >> 16);
}

// A-fragment dword index within a 32-elem (16-dword) K chunk.
__device__ __forceinline__ int adw(int v, int hi) {
  return hi * 8 + v;
}

// Intra-wave LDS producer->consumer sync
__device__ __forceinline__ void wave_lds_sync() {
  asm volatile("s_wait_dscnt 0x0" ::: "memory");
  __builtin_amdgcn_wave_barrier();
}

// CDNA5 async global->LDS copy (ASYNCcnt), 16B per lane
__device__ __forceinline__ void async_ld_b128(unsigned lds_addr, unsigned long long gaddr) {
  asm volatile("global_load_async_to_lds_b128 %0, %1, off"
               :: "v"(lds_addr), "v"(gaddr) : "memory");
}
#define WAIT_ASYNC_0() asm volatile("s_wait_asynccnt 0x0" ::: "memory")
#define WAIT_ASYNC_1() asm volatile("s_wait_asynccnt 0x1" ::: "memory")

// 16-lane (half-wave) reductions
__device__ __forceinline__ float rmax16(float v) {
  v = fmaxf(v, __shfl_xor(v, 1, 32));
  v = fmaxf(v, __shfl_xor(v, 2, 32));
  v = fmaxf(v, __shfl_xor(v, 4, 32));
  v = fmaxf(v, __shfl_xor(v, 8, 32));
  return v;
}
__device__ __forceinline__ float rsum16(float v) {
  v += __shfl_xor(v, 1, 32);
  v += __shfl_xor(v, 2, 32);
  v += __shfl_xor(v, 4, 32);
  v += __shfl_xor(v, 8, 32);
  return v;
}

// ---------------- Kernel 1: f32 -> bf16 convert ----------------
__global__ __launch_bounds__(256) void cvt_bf16_kernel(const float* __restrict__ s,
                                                       unsigned short* __restrict__ d, int n) {
  int i = blockIdx.x * 256 + threadIdx.x;
  if (i < n) d[i] = f2bf(s[i]);
}

// ---------------- Kernel 2: pack weight [K,N] f32 into WMMA B-fragment tiles ----------
__global__ __launch_bounds__(256) void pack_w_kernel(const float* __restrict__ W,
                                                     unsigned short* __restrict__ dst,
                                                     int K, int N) {
  int tid = blockIdx.x * 256 + threadIdx.x;
  int lane = tid & 31;
  int t = tid >> 5;
  int ntn = N >> 4;
  if (t >= (K >> 5) * ntn) return;
  int kt = t / ntn, nt = t - kt * ntn;
  int n = (nt << 4) + (lane & 15);
  int kb = (kt << 5) + ((lane >> 4) << 4);
  __align__(16) unsigned short tmp[16];
#pragma unroll
  for (int e = 0; e < 16; ++e) tmp[e] = f2bf(W[(size_t)(kb + e) * N + n]);
  uint4* o = (uint4*)(dst + (size_t)t * 512 + lane * 16);
  o[0] = ((const uint4*)tmp)[0];
  o[1] = ((const uint4*)tmp)[1];
}

// ---------------- Kernel 3: QKV GEMM  [8192,2048] x [2048,6144] + bias ----------------
// 8 waves, block tile M=256 (32 rows/wave), N=64. B tiles staged to LDS with
// double-buffered async-to-LDS copies (shared by all 8 waves). 8 WMMAs per K-step/wave.
__global__ __launch_bounds__(256) void gemm_qkv_kernel(
    const unsigned short* __restrict__ A, const unsigned short* __restrict__ Wp,
    const float* __restrict__ bias,
    unsigned short* __restrict__ qb, unsigned short* __restrict__ kb2,
    unsigned short* __restrict__ vp) {
  __shared__ __align__(16) unsigned short bsm[2][2048];  // 2 x 4KB B-chunk buffers
  __shared__ float sc[8][16][17];                        // per-wave transpose scratch
  const int KSTEP = H_ / 32;                             // 64
  int tid = threadIdx.x;
  int lane = tid & 31, w = tid >> 5;
  int hi = lane >> 4, nl = lane & 15;
  int Nbase = blockIdx.x * 64;
  int Mbase = blockIdx.y * 256;
  int mw = Mbase + w * 32;
  const unsigned int* Arow0 = (const unsigned int*)(A + (size_t)(mw + nl) * H_);
  const unsigned int* Arow1 = (const unsigned int*)(A + (size_t)(mw + 16 + nl) * H_);
  size_t tcol = (size_t)(Nbase >> 4);
  unsigned bls = (unsigned)(size_t)&bsm[0][0];
  unsigned o16 = (unsigned)tid * 16;

  AccU acc[2][4];
#pragma unroll
  for (int g = 0; g < 2; ++g)
#pragma unroll
    for (int t = 0; t < 4; ++t)
#pragma unroll
      for (int e = 0; e < 8; ++e) acc[g][t].f[e] = 0.f;

  // prologue: stage B chunk 0
  async_ld_b128(bls + o16,
                (unsigned long long)(size_t)(Wp + tcol * 512) + o16);
  for (int j = 0; j < KSTEP; ++j) {
    if (j + 1 < KSTEP) {
      async_ld_b128(bls + ((j + 1) & 1) * 4096 + o16,
                    (unsigned long long)(size_t)(Wp + ((size_t)(j + 1) * (N_QKV >> 4) + tcol) * 512) + o16);
      WAIT_ASYNC_1();
    } else {
      WAIT_ASYNC_0();
    }
    __syncthreads();
    int k0 = j * 32;
    FragAB a0, a1;
#pragma unroll
    for (int v = 0; v < 8; ++v) {
      a0.u[v] = Arow0[(k0 >> 1) + adw(v, hi)];
      a1.u[v] = Arow1[(k0 >> 1) + adw(v, hi)];
    }
    const unsigned short* bbuf = &bsm[j & 1][0];
#pragma unroll
    for (int t = 0; t < 4; ++t) {
      const uint4* bp = (const uint4*)(bbuf + t * 512 + lane * 16);
      FragAB b;
      b.q[0] = bp[0]; b.q[1] = bp[1];
      acc[0][t].v = __builtin_amdgcn_wmma_f32_16x16x32_bf16(false, a0.v, false, b.v,
                                                            (short)0, acc[0][t].v, false, false);
      acc[1][t].v = __builtin_amdgcn_wmma_f32_16x16x32_bf16(false, a1.v, false, b.v,
                                                            (short)0, acc[1][t].v, false, false);
    }
    __syncthreads();
  }

#pragma unroll
  for (int g = 0; g < 2; ++g) {
    int s0 = mw + g * 16;
    int bidx = s0 / SEQ;
    int sl0 = s0 - bidx * SEQ;
#pragma unroll
    for (int t = 0; t < 4; ++t) {
      int n = Nbase + t * 16 + nl;
      float bv = bias[n];
      int which = n / H_;
      int nr = n - which * H_;
      int h = nr >> 7, d = nr & 127;
      size_t bh = (size_t)(bidx * NH_ + h);
      if (which == 0) {               // Q: fold softmax scale
#pragma unroll
        for (int r = 0; r < 8; ++r) {
          float val = (acc[g][t].f[r] + bv) * SCALE_;
          qb[(bh * SEQ + sl0 + r + hi * 8) * HD_ + d] = f2bf(val);
        }
      } else if (which == 1) {        // K row-major per head
#pragma unroll
        for (int r = 0; r < 8; ++r)
          kb2[(bh * SEQ + sl0 + r + hi * 8) * HD_ + d] = f2bf(acc[g][t].f[r] + bv);
      } else {                        // V -> packed B-fragment layout via LDS transpose
#pragma unroll
        for (int r = 0; r < 8; ++r) sc[w][r + hi * 8][nl] = acc[g][t].f[r] + bv;
        wave_lds_sync();
        int hihalf = (sl0 >> 4) & 1;
        int eb = hi * 8;
        __align__(16) unsigned short tmp[8];
#pragma unroll
        for (int jj = 0; jj < 8; ++jj) tmp[jj] = f2bf(sc[w][eb + jj][nl]);
        size_t off = bh * (size_t)(SEQ * HD_)
                   + (size_t)((sl0 >> 5) * (HD_ >> 4) + (d >> 4)) * 512
                   + (size_t)(hihalf * 16 + nl) * 16 + eb;
        *(uint4*)(vp + off) = *(const uint4*)tmp;
        __builtin_amdgcn_wave_barrier();
      }
    }
  }
}

// ---------------- Kernel 4: causal flash attention ------------------------------------
// Block = 8 waves = 8 consecutive 16-row Q tiles of ONE (b,h). K/V chunks (8KB each)
// are staged cooperatively with async-to-LDS; chunk loop is block-uniform so
// __syncthreads is legal; waves skip compute past their causal frontier.
__global__ __launch_bounds__(256) void attn_kernel(
    const unsigned short* __restrict__ qb, const unsigned short* __restrict__ kb,
    const unsigned short* __restrict__ vp, unsigned short* __restrict__ ao) {
  __shared__ __align__(16) unsigned short kt_lds[32 * 136];  // 32 rows x 272B (padded)
  __shared__ __align__(16) unsigned short vt_lds[8 * 512];   // 8 packed tiles x 1KB
  __shared__ __align__(16) unsigned short ps[8][16][34];     // per-wave P transpose
  int tid = threadIdx.x;
  int lane = tid & 31, w = tid >> 5;
  int hi = lane >> 4, nl = lane & 15, hi8 = hi * 8;
  int gw0 = blockIdx.x * 8;
  int bh = gw0 >> 7;
  int it0 = gw0 & 127;
  int it = it0 + w;
  int q0 = it << 4;
  const unsigned short* qB = qb + (size_t)bh * SEQ * HD_;
  const unsigned short* kB = kb + (size_t)bh * SEQ * HD_;
  const unsigned short* vB = vp + (size_t)bh * SEQ * HD_;

  FragAB qf[4];                        // Q 16x128 as 4 A-fragments
  {
    const unsigned int* qrow = (const unsigned int*)(qB + (size_t)(q0 + nl) * HD_);
#pragma unroll
    for (int kt = 0; kt < 4; ++kt)
#pragma unroll
      for (int v = 0; v < 8; ++v) qf[kt].u[v] = qrow[kt * 16 + adw(v, hi)];
  }

  AccU o[8];
#pragma unroll
  for (int dt = 0; dt < 8; ++dt)
#pragma unroll
    for (int e = 0; e < 8; ++e) o[dt].f[e] = 0.f;
  float mrow[8], lrow[8];
#pragma unroll
  for (int r = 0; r < 8; ++r) { mrow[r] = -1e30f; lrow[r] = 0.f; }

  int nch = (q0 + 16 + 31) >> 5;                  // this wave's causal chunk count
  int nch_max = (((it0 + 7) << 4) + 16 + 31) >> 5; // block-uniform chunk count
  unsigned kls = (unsigned)(size_t)kt_lds;
  unsigned vls = (unsigned)(size_t)vt_lds;
  unsigned o16 = (unsigned)tid * 16;

  for (int j = 0; j < nch_max; ++j) {
    // cooperatively stage K chunk (32 keys x 128d, row-padded) and packed V chunk
    {
      unsigned long long gk = (unsigned long long)(size_t)(kB + (size_t)j * 32 * HD_);
      unsigned long long gv = (unsigned long long)(size_t)(vB + (size_t)j * 8 * 512);
      unsigned r0 = o16 >> 8, c0 = o16 & 255u;
      unsigned r1 = (o16 + 4096u) >> 8;
      async_ld_b128(kls + r0 * 272u + c0, gk + o16);
      async_ld_b128(kls + r1 * 272u + c0, gk + o16 + 4096u);
      async_ld_b128(vls + o16, gv + o16);
      async_ld_b128(vls + o16 + 4096u, gv + o16 + 4096u);
      WAIT_ASYNC_0();
    }
    __syncthreads();

    if (j < nch) {
      AccU s[2];
#pragma unroll
      for (int t = 0; t < 2; ++t) {
#pragma unroll
        for (int e = 0; e < 8; ++e) s[t].f[e] = 0.f;
        int rr = t * 16 + nl;        // local key row in staged chunk
#pragma unroll
        for (int kt = 0; kt < 4; ++kt) {
          const uint4* kp = (const uint4*)(kt_lds + rr * 136 + kt * 32 + hi * 16);
          FragAB kf;
          kf.q[0] = kp[0]; kf.q[1] = kp[1];
          s[t].v = __builtin_amdgcn_wmma_f32_16x16x32_bf16(false, qf[kt].v, false, kf.v,
                                                           (short)0, s[t].v, false, false);
        }
      }
      // online softmax (scale folded into Q)
      float scal[8];
#pragma unroll
      for (int r = 0; r < 8; ++r) {
        int qg = q0 + r + hi8;
        int kg0 = j * 32 + nl;
        if (kg0 > qg)      s[0].f[r] = -1e30f;
        if (kg0 + 16 > qg) s[1].f[r] = -1e30f;
        float mx = rmax16(fmaxf(s[0].f[r], s[1].f[r]));
        float mnew = fmaxf(mrow[r], mx);
        scal[r] = __expf(mrow[r] - mnew);
        mrow[r] = mnew;
        float p0 = __expf(s[0].f[r] - mnew);
        float p1 = __expf(s[1].f[r] - mnew);
        s[0].f[r] = p0; s[1].f[r] = p1;
        lrow[r] = lrow[r] * scal[r] + rsum16(p0 + p1);
      }
#pragma unroll
      for (int dt = 0; dt < 8; ++dt)
#pragma unroll
        for (int r = 0; r < 8; ++r) o[dt].f[r] *= scal[r];

      // transpose P -> A-fragment layout through wave-local LDS
#pragma unroll
      for (int r = 0; r < 8; ++r) {
        ps[w][r + hi8][nl]      = f2bf(s[0].f[r]);
        ps[w][r + hi8][16 + nl] = f2bf(s[1].f[r]);
      }
      wave_lds_sync();
      FragAB pf;
      {
        const unsigned int* prow = (const unsigned int*)&ps[w][nl][0];
#pragma unroll
        for (int v = 0; v < 8; ++v) pf.u[v] = prow[adw(v, hi)];
      }
      __builtin_amdgcn_wave_barrier();

      // P (16x32) @ V (32x128) from staged packed tiles
#pragma unroll
      for (int dt = 0; dt < 8; ++dt) {
        const uint4* vq = (const uint4*)(vt_lds + dt * 512 + lane * 16);
        FragAB vf;
        vf.q[0] = vq[0]; vf.q[1] = vq[1];
        o[dt].v = __builtin_amdgcn_wmma_f32_16x16x32_bf16(false, pf.v, false, vf.v,
                                                          (short)0, o[dt].v, false, false);
      }
    }
    __syncthreads();
  }

  // normalize and store merged-head output, row-major bf16 [B*S, H]
  int b = bh >> 4, h = bh & 15;
#pragma unroll
  for (int dt = 0; dt < 8; ++dt)
#pragma unroll
    for (int r = 0; r < 8; ++r) {
      float val = o[dt].f[r] / lrow[r];
      ao[(size_t)(b * SEQ + q0 + r + hi8) * H_ + h * HD_ + dt * 16 + nl] = f2bf(val);
    }
}

// ---------------- Kernel 5: output projection [8192,2048] x [2048,2048] + bias --------
__global__ __launch_bounds__(256) void gemm_proj_kernel(
    const unsigned short* __restrict__ A, const unsigned short* __restrict__ Wp,
    const float* __restrict__ bias, float* __restrict__ out) {
  __shared__ __align__(16) unsigned short bsm[2][2048];
  const int KSTEP = H_ / 32;
  int tid = threadIdx.x;
  int lane = tid & 31, w = tid >> 5;
  int hi = lane >> 4, nl = lane & 15;
  int Nbase = blockIdx.x * 64;
  int Mbase = blockIdx.y * 256;
  int mw = Mbase + w * 32;
  const unsigned int* Arow0 = (const unsigned int*)(A + (size_t)(mw + nl) * H_);
  const unsigned int* Arow1 = (const unsigned int*)(A + (size_t)(mw + 16 + nl) * H_);
  size_t tcol = (size_t)(Nbase >> 4);
  unsigned bls = (unsigned)(size_t)&bsm[0][0];
  unsigned o16 = (unsigned)tid * 16;

  AccU acc[2][4];
#pragma unroll
  for (int g = 0; g < 2; ++g)
#pragma unroll
    for (int t = 0; t < 4; ++t)
#pragma unroll
      for (int e = 0; e < 8; ++e) acc[g][t].f[e] = 0.f;

  async_ld_b128(bls + o16,
                (unsigned long long)(size_t)(Wp + tcol * 512) + o16);
  for (int j = 0; j < KSTEP; ++j) {
    if (j + 1 < KSTEP) {
      async_ld_b128(bls + ((j + 1) & 1) * 4096 + o16,
                    (unsigned long long)(size_t)(Wp + ((size_t)(j + 1) * (H_ >> 4) + tcol) * 512) + o16);
      WAIT_ASYNC_1();
    } else {
      WAIT_ASYNC_0();
    }
    __syncthreads();
    int k0 = j * 32;
    FragAB a0, a1;
#pragma unroll
    for (int v = 0; v < 8; ++v) {
      a0.u[v] = Arow0[(k0 >> 1) + adw(v, hi)];
      a1.u[v] = Arow1[(k0 >> 1) + adw(v, hi)];
    }
    const unsigned short* bbuf = &bsm[j & 1][0];
#pragma unroll
    for (int t = 0; t < 4; ++t) {
      const uint4* bp = (const uint4*)(bbuf + t * 512 + lane * 16);
      FragAB b;
      b.q[0] = bp[0]; b.q[1] = bp[1];
      acc[0][t].v = __builtin_amdgcn_wmma_f32_16x16x32_bf16(false, a0.v, false, b.v,
                                                            (short)0, acc[0][t].v, false, false);
      acc[1][t].v = __builtin_amdgcn_wmma_f32_16x16x32_bf16(false, a1.v, false, b.v,
                                                            (short)0, acc[1][t].v, false, false);
    }
    __syncthreads();
  }
#pragma unroll
  for (int g = 0; g < 2; ++g)
#pragma unroll
    for (int t = 0; t < 4; ++t) {
      int n = Nbase + t * 16 + nl;
      float bv = bias[n];
#pragma unroll
      for (int r = 0; r < 8; ++r)
        out[(size_t)(mw + g * 16 + r + hi * 8) * H_ + n] = acc[g][t].f[r] + bv;
    }
}

// --------------------------------- launch ---------------------------------------------
extern "C" void kernel_launch(void* const* d_in, const int* in_sizes, int n_in,
                              void* d_out, int out_size, void* d_ws, size_t ws_size,
                              hipStream_t stream) {
  (void)in_sizes; (void)n_in; (void)out_size; (void)ws_size;
  const float* hs  = (const float*)d_in[0];   // [4,2048,2048]
  const float* Wat = (const float*)d_in[1];   // [2048,6144]
  const float* bat = (const float*)d_in[2];   // [6144]
  const float* Wpr = (const float*)d_in[3];   // [2048,2048]
  const float* bpr = (const float*)d_in[4];   // [2048]
  float* out = (float*)d_out;                 // [4,2048,2048] f32

  char* p = (char*)d_ws;
  unsigned short* hsb   = (unsigned short*)p; p += (size_t)M_TOT * H_ * 2;
  unsigned short* wat_p = (unsigned short*)p; p += (size_t)H_ * N_QKV * 2;
  unsigned short* wpr_p = (unsigned short*)p; p += (size_t)H_ * H_ * 2;
  unsigned short* qb    = (unsigned short*)p; p += (size_t)M_TOT * H_ * 2;
  unsigned short* kbuf  = (unsigned short*)p; p += (size_t)M_TOT * H_ * 2;
  unsigned short* vpk   = (unsigned short*)p; p += (size_t)M_TOT * H_ * 2;
  unsigned short* ao    = (unsigned short*)p; p += (size_t)M_TOT * H_ * 2;

  {
    int n = M_TOT * H_;
    cvt_bf16_kernel<<<(n + 255) / 256, 256, 0, stream>>>(hs, hsb, n);
  }
  {
    int thr = (H_ / 32) * (N_QKV / 16) * 32;
    pack_w_kernel<<<(thr + 255) / 256, 256, 0, stream>>>(Wat, wat_p, H_, N_QKV);
  }
  {
    int thr = (H_ / 32) * (H_ / 16) * 32;
    pack_w_kernel<<<(thr + 255) / 256, 256, 0, stream>>>(Wpr, wpr_p, H_, H_);
  }
  gemm_qkv_kernel<<<dim3(N_QKV / 64, M_TOT / 256), 256, 0, stream>>>(hsb, wat_p, bat,
                                                                     qb, kbuf, vpk);
  attn_kernel<<<(BATCH * NH_ * (SEQ / 16)) / 8, 256, 0, stream>>>(qb, kbuf, vpk, ao);
  gemm_proj_kernel<<<dim3(H_ / 64, M_TOT / 256), 256, 0, stream>>>(ao, wpr_p, bpr, out);
}
